// Model_55241869361850
// MI455X (gfx1250) — matrix operand order repaired and checked by reading, not compiled, not run on hardware
//
#include <hip/hip_runtime.h>

typedef __attribute__((ext_vector_type(16))) _Float16 v16h;
typedef __attribute__((ext_vector_type(4)))  _Float16 h4;
typedef __attribute__((ext_vector_type(8)))  float    v8f;

#define TAU 18.0f

static constexpr int Nn = 2048, Pp = 1024, Dd = 256;

// Output offsets (floats), concatenated in reference return order
static constexpr long OFF_XHAT  = 0;
static constexpr long OFF_YHAT  = (long)Nn * Dd;
static constexpr long OFF_XBOX  = 2L * Nn * Dd;
static constexpr long OFF_YBOX  = 3L * Nn * Dd;
static constexpr long OFF_YPRED = 4L * Nn * Dd;
static constexpr long OFF_SREC  = 5L * Nn * Dd;
static constexpr long OFF_TREC  = OFF_SREC + (long)Nn * Pp;
static constexpr long OFF_TPRED = OFF_TREC + (long)Nn * Pp;

// -------------------------------------------------------------------------
// v[i][j] = exp(-tau * sigmoid(rawM[i][j]))
// -------------------------------------------------------------------------
__global__ __launch_bounds__(256) void vmat_kernel(const float* __restrict__ rawM,
                                                   float* __restrict__ V)
{
    int idx = blockIdx.x * 256 + threadIdx.x;
    if (idx < Dd * Dd) {
        float x   = rawM[idx];
        float sig = __builtin_amdgcn_rcpf(1.0f + __expf(-x));
        V[idx]    = __expf(-TAU * sig);
    }
}

// -------------------------------------------------------------------------
// Staging helpers: block tile = 128(M) x 32(K) for A, 32(K) x 64(N) for B.
// All loads go to distinct registers first (overlappable), stores convert
// fp32 -> f16 and write b64 chunks into the padded LDS tiles.
// -------------------------------------------------------------------------
__device__ __forceinline__ void stage_load(const float* __restrict__ Ag, int lda,
                                           const float* __restrict__ Bg, int ldb,
                                           int rowA0, int colB0, int k0, int tid,
                                           float4 (&aR)[8], float (&bR)[4][4])
{
    #pragma unroll
    for (int q = 0; q < 8; ++q) {                 // A: 128x32 = 1024 float4
        int s  = tid + q * 128;
        int r  = s >> 3;                          // 0..127
        int cv = (s & 7) * 4;                     // 0..28
        aR[q] = *(const float4*)(Ag + (long)(rowA0 + r) * lda + k0 + cv);
    }
    #pragma unroll
    for (int q = 0; q < 4; ++q) {                 // B: columns of 4 K for fixed N
        int s  = tid + q * 128;                   // 0..511
        int n  = s & 63;
        int kg = s >> 6;                          // 0..7
        #pragma unroll
        for (int c = 0; c < 4; ++c)
            bR[q][c] = Bg[(long)(k0 + kg * 4 + c) * ldb + colB0 + n];
    }
}

__device__ __forceinline__ void stage_store(_Float16 (*As)[48], _Float16 (*BsT)[48],
                                            int tid,
                                            const float4 (&aR)[8], const float (&bR)[4][4])
{
    #pragma unroll
    for (int q = 0; q < 8; ++q) {
        int s  = tid + q * 128;
        int r  = s >> 3;
        int cv = (s & 7) * 4;
        h4 h = { (_Float16)aR[q].x, (_Float16)aR[q].y,
                 (_Float16)aR[q].z, (_Float16)aR[q].w };
        *(h4*)&As[r][cv] = h;                     // b64 store, 8B aligned
    }
    #pragma unroll
    for (int q = 0; q < 4; ++q) {
        int s  = tid + q * 128;
        int n  = s & 63;
        int kg = s >> 6;
        h4 h = { (_Float16)bR[q][0], (_Float16)bR[q][1],
                 (_Float16)bR[q][2], (_Float16)bR[q][3] };
        *(h4*)&BsT[n][kg * 4] = h;                // b64 store, 8B aligned
    }
}

// -------------------------------------------------------------------------
// WMMA GEMM core: block tile 128x64, 128 threads = 4 waves.
// Wave w owns rows 32*w..32*w+31 (2 m-subtiles) x all 64 cols: 8 accumulators,
// 8 v_wmma_f32_16x16x32_f16 per 32-K step. LDS double-buffered.
// -------------------------------------------------------------------------
__device__ __forceinline__ void gemm_core(const float* __restrict__ Ag, int lda,
                                          const float* __restrict__ Bg, int ldb,
                                          int K, int mBlk, int nBlk,
                                          float* __restrict__ Cg, int ldc,
                                          float* __restrict__ CboxG)
{
    __shared__ __align__(16) _Float16 As [2][128][48];   // [buf][m][k]  (96B rows)
    __shared__ __align__(16) _Float16 BsT[2][ 64][48];   // [buf][n][k]

    const int tid  = threadIdx.x;
    const int wave = tid >> 5;
    const int lane = tid & 31;
    const int half = lane >> 4;
    const int l16  = lane & 15;

    const int rowA0 = mBlk * 128;
    const int colB0 = nBlk * 64;
    const int KT    = K >> 5;

    v8f acc[2][4];
    #pragma unroll
    for (int ms = 0; ms < 2; ++ms)
        #pragma unroll
        for (int t = 0; t < 4; ++t) acc[ms][t] = (v8f){};

    float4 aR[8];
    float  bR[4][4];

    stage_load(Ag, lda, Bg, ldb, rowA0, colB0, 0, tid, aR, bR);
    stage_store(As[0], BsT[0], tid, aR, bR);
    __syncthreads();

    for (int kt = 0; kt < KT; ++kt) {
        const int cur = kt & 1;
        if (kt + 1 < KT)   // overlap next tile's global loads with this tile's WMMAs
            stage_load(Ag, lda, Bg, ldb, rowA0, colB0, (kt + 1) * 32, tid, aR, bR);

        // A fragments (16x32 f16): lanes<16 hold K {0..7,16..23}, lanes>=16 {8..15,24..31}
        v16h af[2];
        #pragma unroll
        for (int ms = 0; ms < 2; ++ms) {
            const int mrow = 32 * wave + 16 * ms + l16;
            #pragma unroll
            for (int i = 0; i < 8; ++i) {
                int kk = ((i < 4) ? 2 * i : 2 * i + 8) + 8 * half;
                af[ms][2 * i + 0] = As[cur][mrow][kk + 0];
                af[ms][2 * i + 1] = As[cur][mrow][kk + 1];
            }
        }
        #pragma unroll
        for (int t = 0; t < 4; ++t) {
            v16h bf;   // lanes<16: K {0..7,16..23}, lanes>=16: K {8..15,24..31}, N = l16
            #pragma unroll
            for (int i = 0; i < 8; ++i) {
                int kk = ((i < 4) ? 2 * i : 2 * i + 8) + 8 * half;
                bf[2 * i + 0] = BsT[cur][t * 16 + l16][kk + 0];
                bf[2 * i + 1] = BsT[cur][t * 16 + l16][kk + 1];
            }
            #pragma unroll
            for (int ms = 0; ms < 2; ++ms)
                acc[ms][t] = __builtin_amdgcn_wmma_f32_16x16x32_f16(
                                 false, af[ms], false, bf, (short)0, acc[ms][t],
                                 false, false);
        }

        if (kt + 1 < KT)
            stage_store(As[cur ^ 1], BsT[cur ^ 1], tid, aR, bR);
        __syncthreads();
    }

    // Epilogue: C/D layout: element r -> row (r + 8*half), col l16
    #pragma unroll
    for (int ms = 0; ms < 2; ++ms) {
        #pragma unroll
        for (int t = 0; t < 4; ++t) {
            #pragma unroll
            for (int r = 0; r < 8; ++r) {
                int  row = rowA0 + 32 * wave + 16 * ms + r + 8 * half;
                int  col = colB0 + t * 16 + l16;
                float v  = acc[ms][t][r];
                Cg[(long)row * ldc + col] = v;
                if (CboxG)
                    CboxG[(long)row * ldc + col] = fminf(fmaxf(v, 0.0f), 1.0f);
            }
        }
    }
}

// GEMM 1: z=0 -> X_hat/X_box = S@A ; z=1 -> Y_hat/Y_box = T@A
__global__ __launch_bounds__(128) void gemm1_kernel(const float* __restrict__ S,
                                                    const float* __restrict__ T,
                                                    const float* __restrict__ A,
                                                    float* out)
{
    const float* src = (blockIdx.z == 0) ? S : T;
    float* hat = out + ((blockIdx.z == 0) ? OFF_XHAT : OFF_YHAT);
    float* box = out + ((blockIdx.z == 0) ? OFF_XBOX : OFF_YBOX);
    gemm_core(src, Pp, A, Dd, Pp, blockIdx.x, blockIdx.y, hat, Dd, box);
}

// GEMM 2: {X_hat,Y_hat,Y_pred} @ B -> {S_rec,T_rec,T_pred}
__global__ __launch_bounds__(128) void gemm2_kernel(const float* __restrict__ B,
                                                    float* out)
{
    const float* src;
    float*       dst;
    if      (blockIdx.z == 0) { src = out + OFF_XHAT;  dst = out + OFF_SREC;  }
    else if (blockIdx.z == 1) { src = out + OFF_YHAT;  dst = out + OFF_TREC;  }
    else                      { src = out + OFF_YPRED; dst = out + OFF_TPRED; }
    gemm_core(src, Dd, B, Pp, Dd, blockIdx.x, blockIdx.y, dst, Pp, nullptr);
}

// -------------------------------------------------------------------------
// Y_pred[n,j] = (1/tau) * log( sum_i 1 / (u[n,i] + v[i,j]) )
// u[n,i] = exp(-tau * X_box[n,i]).  16 rows per block, thread j owns column j.
// u tile read back as float4 (one ds_load_b128 feeds 4 rcp+adds).
// -------------------------------------------------------------------------
__global__ __launch_bounds__(256) void compose_kernel(const float* __restrict__ xbox,
                                                      const float* __restrict__ V,
                                                      float* __restrict__ ypred)
{
    __shared__ __align__(16) float us[16][260];   // 1040B rows, 16B aligned
    const int j  = threadIdx.x;
    const int n0 = blockIdx.x * 16;

    #pragma unroll
    for (int r = 0; r < 16; ++r)
        us[r][j] = __expf(-TAU * xbox[(long)(n0 + r) * Dd + j]);
    __syncthreads();

    float acc[16];
    #pragma unroll
    for (int r = 0; r < 16; ++r) acc[r] = 0.0f;

    for (int i0 = 0; i0 < Dd; i0 += 4) {
        float vv0 = V[(i0 + 0) * Dd + j];         // coalesced, L2-resident
        float vv1 = V[(i0 + 1) * Dd + j];
        float vv2 = V[(i0 + 2) * Dd + j];
        float vv3 = V[(i0 + 3) * Dd + j];
        #pragma unroll
        for (int r = 0; r < 16; ++r) {
            const float4 u4 = *(const float4*)&us[r][i0];  // broadcast b128
            acc[r] += __builtin_amdgcn_rcpf(u4.x + vv0);
            acc[r] += __builtin_amdgcn_rcpf(u4.y + vv1);
            acc[r] += __builtin_amdgcn_rcpf(u4.z + vv2);
            acc[r] += __builtin_amdgcn_rcpf(u4.w + vv3);
        }
    }

    #pragma unroll
    for (int r = 0; r < 16; ++r)
        ypred[(long)(n0 + r) * Dd + j] = (1.0f / TAU) * __logf(acc[r]);
}

// -------------------------------------------------------------------------
extern "C" void kernel_launch(void* const* d_in, const int* in_sizes, int n_in,
                              void* d_out, int out_size, void* d_ws, size_t ws_size,
                              hipStream_t stream)
{
    (void)in_sizes; (void)n_in; (void)out_size; (void)ws_size;

    const float* S    = (const float*)d_in[0];
    const float* T    = (const float*)d_in[1];
    const float* A    = (const float*)d_in[2];
    const float* rawM = (const float*)d_in[3];
    const float* B    = (const float*)d_in[4];
    float* out = (float*)d_out;
    float* V   = (float*)d_ws;                 // 256 KB scratch

    vmat_kernel   <<<dim3((Dd * Dd + 255) / 256), 256, 0, stream>>>(rawM, V);
    gemm1_kernel  <<<dim3(Nn / 128, Dd / 64, 2), 128, 0, stream>>>(S, T, A, out);
    compose_kernel<<<dim3(Nn / 16), 256, 0, stream>>>(out + OFF_XBOX, V, out + OFF_YPRED);
    gemm2_kernel  <<<dim3(Nn / 128, Pp / 64, 3), 128, 0, stream>>>(B, out);
}
